// SAGE_Dist_2233382994520
// MI455X (gfx1250) — matrix operand order repaired and checked by reading, not compiled, not run on hardware
//
#include <hip/hip_runtime.h>

typedef __attribute__((ext_vector_type(16))) _Float16 v16h;
typedef __attribute__((ext_vector_type(8)))  _Float16 v8h;
typedef __attribute__((ext_vector_type(4)))  _Float16 v4h;
typedef __attribute__((ext_vector_type(8)))  float    v8f;
typedef __attribute__((ext_vector_type(4)))  float    v4f;

#define DIM 128

// ---------------- degree ----------------
__global__ void k_deg(const long long* __restrict__ ei, float* __restrict__ deg, int E) {
  int t = blockIdx.x * blockDim.x + threadIdx.x;
  if (t < E) atomicAdd(&deg[(int)ei[(long long)E + t]], 1.0f);
}

__global__ void k_invdeg(float* __restrict__ deg, int n) {
  int t = blockIdx.x * blockDim.x + threadIdx.x;
  if (t < n) deg[t] = 1.0f / fmaxf(deg[t], 1.0f);
}

// ---------------- f32 -> f16 activation conversion (4 elems/thread) ----------------
__global__ void k_cvt(const float* __restrict__ x, _Float16* __restrict__ h, int n4) {
  int t = blockIdx.x * blockDim.x + threadIdx.x;
  if (t >= n4) return;
  v4f v = *(const v4f*)(x + (long long)t * 4);
  v4h o;
  o[0] = (_Float16)v[0]; o[1] = (_Float16)v[1];
  o[2] = (_Float16)v[2]; o[3] = (_Float16)v[3];
  *(v4h*)(h + (long long)t * 4) = o;
}

// ---------------- weight packing into WMMA B-fragment layout ----------------
// dst[((ntile*4+ks)*32 + lane)*16 + j] = W[n,k], n = ntile*16 + (lane&15),
// k = ks*32 + (lane<16 ? 0 : 8) + (j<8 ? j : j+8)   (B is W^T; lane holds K in two 8-wide chunks)
__global__ void k_pack(const float* __restrict__ W, _Float16* __restrict__ dst) {
  int t = blockIdx.x * 256 + threadIdx.x;          // 16384 threads per matrix
  int j    = t & 15;
  int lane = (t >> 4) & 31;
  int frag = t >> 9;                               // ntile*4 + ks
  int ntile = frag >> 2, ks = frag & 3;
  int n = ntile * 16 + (lane & 15);
  int k = ks * 32 + ((lane < 16) ? 0 : 8) + ((j < 8) ? j : j + 8);
  dst[t] = (_Float16)W[n * 128 + k];
}

// ---------------- edge aggregation: agg[dst] += h[src] (f16 gather, f32 atomics) ----------------
__global__ void k_agg(const _Float16* __restrict__ h, const long long* __restrict__ ei,
                      float* __restrict__ agg, int E) {
  long long gid = (long long)blockIdx.x * blockDim.x + threadIdx.x;
  int e = (int)(gid >> 5);
  if (e >= E) return;
  int part = (int)gid & 31;                        // 32 lanes/edge, 4 f16 each
  int s = (int)ei[e];
  int d = (int)ei[(long long)E + e];
  v4h v = *(const v4h*)(h + (long long)s * DIM + part * 4);
  float* ap = agg + (long long)d * DIM + part * 4;
  atomicAdd(ap + 0, (float)v[0]);
  atomicAdd(ap + 1, (float)v[1]);
  atomicAdd(ap + 2, (float)v[2]);
  atomicAdd(ap + 3, (float)v[3]);
}

// ---------------- fused SAGE layer GEMM: D = (agg*invdeg)@Wl^T + h@Wr^T + bl, [relu] ----------------
template <int RELU, int OUT_F16>
__global__ void k_gemm(const float* __restrict__ agg, const _Float16* __restrict__ h,
                       const _Float16* __restrict__ pWl, const _Float16* __restrict__ pWr,
                       const float* __restrict__ bias, const float* __restrict__ invdeg,
                       _Float16* __restrict__ outh, float* __restrict__ outf, int ntiles) {
  extern __shared__ char smem_raw[];               // 131072 B: Wl frags then Wr frags
  _Float16* smem = (_Float16*)smem_raw;

  // Stage both packed weight matrices into LDS (ds path, saves L2 re-reads per wave)
  {
    const uint4* sl = (const uint4*)pWl;
    const uint4* sr = (const uint4*)pWr;
    uint4* dv = (uint4*)smem_raw;
    for (int i = threadIdx.x; i < 4096; i += 256) dv[i] = sl[i];
    for (int i = threadIdx.x; i < 4096; i += 256) dv[4096 + i] = sr[i];
  }
  __syncthreads();

  int wave = threadIdx.x >> 5;
  int lane = threadIdx.x & 31;
  int mtile = blockIdx.x * 8 + wave;
  if (mtile >= ntiles) return;                     // whole-wave exit: EXEC stays all-ones

  long long row = (long long)mtile * 16 + (lane & 15);   // A-matrix: M = lane&15
  int k0 = (lane < 16) ? 0 : 8;                    // per-lane K base within 32-wide k-step
  float invd = invdeg[row];

  const float*    ap = agg + row * DIM;
  const _Float16* hp = h   + row * DIM;

  v16h aAgg[4], aH[4];
#pragma unroll
  for (int ks = 0; ks < 4; ++ks) {
    int base = ks * 32 + k0;
    // agg: 8+8 f32, scale by invdeg, convert to f16 fragment
    v4f f0 = *(const v4f*)(ap + base);
    v4f f1 = *(const v4f*)(ap + base + 4);
    v4f f2 = *(const v4f*)(ap + base + 16);
    v4f f3 = *(const v4f*)(ap + base + 20);
    v16h a;
#pragma unroll
    for (int i = 0; i < 4; ++i) {
      a[i]      = (_Float16)(f0[i] * invd);
      a[4 + i]  = (_Float16)(f1[i] * invd);
      a[8 + i]  = (_Float16)(f2[i] * invd);
      a[12 + i] = (_Float16)(f3[i] * invd);
    }
    aAgg[ks] = a;
    // h: two contiguous 16B f16 chunks
    v8h lo = *(const v8h*)(hp + base);
    v8h hi = *(const v8h*)(hp + base + 16);
    aH[ks] = __builtin_shufflevector(lo, hi, 0,1,2,3,4,5,6,7,8,9,10,11,12,13,14,15);
  }

#pragma unroll
  for (int ntile = 0; ntile < 8; ++ntile) {
    float bval = bias[ntile * 16 + (lane & 15)];   // C/D: N = lane&15
    v8f c;
#pragma unroll
    for (int i = 0; i < 8; ++i) c[i] = bval;

    const _Float16* wl = smem +         (ntile * 4) * 512 + lane * 16;
    const _Float16* wr = smem + 32768 + (ntile * 4) * 512 + lane * 16;

#pragma unroll
    for (int ks = 0; ks < 4; ++ks) {
      v8h b0 = *(const v8h*)(wl + ks * 512);
      v8h b1 = *(const v8h*)(wl + ks * 512 + 8);
      v16h bf = __builtin_shufflevector(b0, b1, 0,1,2,3,4,5,6,7,8,9,10,11,12,13,14,15);
      c = __builtin_amdgcn_wmma_f32_16x16x32_f16(false, aAgg[ks], false, bf, (short)0, c, false, false);
    }
#pragma unroll
    for (int ks = 0; ks < 4; ++ks) {
      v8h b0 = *(const v8h*)(wr + ks * 512);
      v8h b1 = *(const v8h*)(wr + ks * 512 + 8);
      v16h bf = __builtin_shufflevector(b0, b1, 0,1,2,3,4,5,6,7,8,9,10,11,12,13,14,15);
      c = __builtin_amdgcn_wmma_f32_16x16x32_f16(false, aH[ks], false, bf, (short)0, c, false, false);
    }

    // D layout: lane writes 8 rows; M = r + 8*(lane>=16), N = lane&15
    int ncol  = ntile * 16 + (lane & 15);
    int mbase = mtile * 16 + ((lane < 16) ? 0 : 8);
#pragma unroll
    for (int r = 0; r < 8; ++r) {
      float v = c[r];
      if (RELU) v = fmaxf(v, 0.0f);
      long long idx = (long long)(mbase + r) * DIM + ncol;
      if (OUT_F16) outh[idx] = (_Float16)v;
      else         outf[idx] = v;
    }
  }
}

extern "C" void kernel_launch(void* const* d_in, const int* in_sizes, int n_in,
                              void* d_out, int out_size, void* d_ws, size_t ws_size,
                              hipStream_t stream) {
  const float*     x   = (const float*)d_in[0];
  const long long* ei  = (const long long*)d_in[1];
  const float* Wl0 = (const float*)d_in[2];
  const float* bl0 = (const float*)d_in[3];
  const float* Wr0 = (const float*)d_in[4];
  const float* Wl1 = (const float*)d_in[5];
  const float* bl1 = (const float*)d_in[6];
  const float* Wr1 = (const float*)d_in[7];
  const float* Wl2 = (const float*)d_in[8];
  const float* bl2 = (const float*)d_in[9];
  const float* Wr2 = (const float*)d_in[10];

  const int E = in_sizes[1] / 2;
  const int N = in_sizes[0] / DIM;                 // 100000 (divisible by 16)
  const int NT = N / 16;                           // 6250 row tiles

  // workspace layout
  char* ws = (char*)d_ws;
  size_t off = 0;
  auto take = [&](size_t bytes) { char* p = ws + off; off += (bytes + 255) & ~(size_t)255; return p; };
  float*    invdeg = (float*)take((size_t)N * 4);
  _Float16* bufA   = (_Float16*)take((size_t)N * DIM * 2);
  _Float16* bufB   = (_Float16*)take((size_t)N * DIM * 2);
  float*    agg    = (float*)take((size_t)N * DIM * 4);
  _Float16* pw     = (_Float16*)take((size_t)6 * 128 * 128 * 2);
  _Float16 *pWl0 = pw,          *pWr0 = pw + 16384;
  _Float16 *pWl1 = pw + 32768,  *pWr1 = pw + 49152;
  _Float16 *pWl2 = pw + 65536,  *pWr2 = pw + 81920;

  // degree -> 1/max(deg,1)
  hipMemsetAsync(invdeg, 0, (size_t)N * 4, stream);
  k_deg<<<(E + 255) / 256, 256, 0, stream>>>(ei, invdeg, E);
  k_invdeg<<<(N + 255) / 256, 256, 0, stream>>>(invdeg, N);

  // f16 activations + packed f16 weights
  k_cvt<<<(N * DIM / 4 + 255) / 256, 256, 0, stream>>>(x, bufA, N * DIM / 4);
  k_pack<<<64, 256, 0, stream>>>(Wl0, pWl0);
  k_pack<<<64, 256, 0, stream>>>(Wr0, pWr0);
  k_pack<<<64, 256, 0, stream>>>(Wl1, pWl1);
  k_pack<<<64, 256, 0, stream>>>(Wr1, pWr1);
  k_pack<<<64, 256, 0, stream>>>(Wl2, pWl2);
  k_pack<<<64, 256, 0, stream>>>(Wr2, pWr2);

  const int gridAgg  = (int)(((long long)E * 32 + 255) / 256);
  const int gridGemm = (NT + 7) / 8;
  const size_t ldsB  = 2 * 128 * 128 * 2;          // 64 KB (of 320 KB/WGP)

  // layer 0: x -> h1 (relu)
  hipMemsetAsync(agg, 0, (size_t)N * DIM * 4, stream);
  k_agg<<<gridAgg, 256, 0, stream>>>(bufA, ei, agg, E);
  k_gemm<1, 1><<<gridGemm, 256, ldsB, stream>>>(agg, bufA, pWl0, pWr0, bl0, invdeg, bufB, nullptr, NT);

  // layer 1: h1 -> h2 (relu)
  hipMemsetAsync(agg, 0, (size_t)N * DIM * 4, stream);
  k_agg<<<gridAgg, 256, 0, stream>>>(bufB, ei, agg, E);
  k_gemm<1, 1><<<gridGemm, 256, ldsB, stream>>>(agg, bufB, pWl1, pWr1, bl1, invdeg, bufA, nullptr, NT);

  // layer 2: h2 -> out (f32, no relu)
  hipMemsetAsync(agg, 0, (size_t)N * DIM * 4, stream);
  k_agg<<<gridAgg, 256, 0, stream>>>(bufA, ei, agg, E);
  k_gemm<0, 0><<<gridGemm, 256, ldsB, stream>>>(agg, bufA, pWl2, pWr2, bl2, invdeg, nullptr, (float*)d_out, NT);
}
